// Kagome108RegularConv2DV2_20633022890676
// MI455X (gfx1250) — matrix operand order repaired and checked by reading, not compiled, not run on hardware
//
#include <hip/hip_runtime.h>
#include <cstdint>
#include <cstddef>

// ---- CDNA5 (gfx1250) types -------------------------------------------------
typedef __attribute__((ext_vector_type(16))) __bf16 v16bf;
typedef __attribute__((ext_vector_type(8)))  __bf16 v8bf;
typedef __attribute__((ext_vector_type(8)))  float  v8f;

#define NPOS      324           // 18*18 padded lattice positions
#define ROWB      272           // bytes per LDS row: 128ch*2B data + 16B bank-skew pad
#define LDS_BYTES (NPOS * ROWB) // 88,128 B dynamic LDS per workgroup (WGP has 320KB)

// Boundary-condition copy table: padded[dst] = padded[src] (src always interior).
__constant__ unsigned char cDR[37] = {1,1,2,3,4,4,6,7,8,10,11,12,14,14,15,16,17,17,16,15,14,14,12,10,8,6,4,4,3,2,5,9,13,15,17,15,13};
__constant__ unsigned char cDC[37] = {3,5,7,9,10,11,13,13,14,15,15,16,15,16,15,14,13,11,9,7,6,5,3,2,1,0,0,1,1,2,0,2,4,8,12,14,16};
__constant__ unsigned char cSR[37] = {13,13,14,15,16,16,6,7,8,10,11,12,2,2,3,4,5,5,4,3,2,2,12,10,8,6,16,16,15,14,5,9,1,3,5,3,1};
__constant__ unsigned char cSC[37] = {15,5,7,9,10,11,1,1,2,3,3,4,3,4,3,2,1,11,9,7,6,5,15,14,13,12,12,13,13,14,12,14,4,8,12,2,4};

// Output mask, one bit per column, precomputed from the reference's _make_mask().
__constant__ unsigned short cMASK[16] = {0x0008,0x003C,0x00FE,0x01FE,0x0FFF,0x0FFF,0x0FFF,0x1FFE,
                                         0x3FFC,0x3FFC,0x3FFC,0x7FF8,0x7FF0,0x3FC0,0x1F00,0x1E00};

__device__ __forceinline__ unsigned short f32_to_bf16_rne(float f) {
    unsigned int u = __float_as_uint(f);
    return (unsigned short)((u + 0x7FFFu + ((u >> 16) & 1u)) >> 16);
}

// ---- Weight repack: W[co][ci][3][3] f32 -> Wb[co][tap][ci] bf16 ------------
__global__ void __launch_bounds__(256)
pack_weights_kernel(const float* __restrict__ W, unsigned short* __restrict__ Wb) {
    int idx = blockIdx.x * 256 + threadIdx.x;
    if (idx >= 128 * 1152) return;
    int co = idx / 1152;
    int r  = idx - co * 1152;
    int t  = r >> 7;            // tap = kh*3+kw
    int ci = r & 127;
    Wb[idx] = f32_to_bf16_rne(W[(co * 128 + ci) * 9 + t]);
}

// ---- Fused pad+BC+conv+bias+mask, WMMA implicit GEMM ------------------------
// grid.x = B (one image per workgroup), 256 threads = 8 wave32.
// Wave w computes Cout tile [16w,16w+16); N tiles are the 16 output rows.
__global__ void __launch_bounds__(256)
kagome_conv_wmma_kernel(const float* __restrict__ x,
                        const unsigned short* __restrict__ Wb,
                        const float* __restrict__ bias,
                        float* __restrict__ out) {
    extern __shared__ char smem[];

    const int b    = blockIdx.x;
    const int tid  = threadIdx.x;
    const int wave = tid >> 5;
    const int lane = tid & 31;
    const int m    = lane & 15;   // A row (co within tile) == B column (output col w)
    const int hi   = lane >> 4;   // K-half select per ISA A/B VGPR layouts

    // ---------- Stage 1a: interior fill, coalesced 64B row reads --------------
    // Segment s = c*16 + r : half-wave per segment, lanes read 16 consecutive
    // floats of x[b][c][r][:], convert to bf16, scatter into LDS [pos][ch]
    // (272B row skew -> b16 scatters spread across banks).
    {
        const float* xb = x + (((size_t)b * 128) << 8);
        for (int s = (wave << 1) + hi; s < 2048; s += 16) {
            const int c = s >> 4;
            const int r = s & 15;               // x row; padded row = r+1
            const float v = xb[((size_t)c << 8) + (r << 4) + m];
            *(unsigned short*)(smem + ((r + 1) * 18 + m + 1) * ROWB + (c << 1)) =
                f32_to_bf16_rne(v);
        }
    }

    // ---------- Stage 1b: zero the 68 border positions (disjoint from 1a) -----
    for (int i = tid; i < 68 * 64; i += 256) {
        const int bi = i >> 6, w = i & 63;
        int p;
        if (bi < 18)        p = bi;                                  // row 0
        else if (bi < 36)   p = 306 + (bi - 18);                     // row 17
        else {                                                       // cols 0/17
            const int r = 1 + ((bi - 36) >> 1);
            p = r * 18 + ((bi & 1) ? 17 : 0);
        }
        *(unsigned int*)(smem + p * ROWB + (w << 2)) = 0u;
    }
    __syncthreads();

    // ---------- Stage 1c: boundary-condition overwrites (37 positions) --------
    // Sources are interior => values come straight from x (matches the
    // reference's single pre-copy gather). Two channels packed per b32 store.
    for (int i = wave; i < 37; i += 8) {
        const int sr = cSR[i], sc = cSC[i];
        const int p  = (int)cDR[i] * 18 + (int)cDC[i];
        const float* xs = x + (((size_t)b * 128) << 8) + ((sr - 1) << 4) + (sc - 1);
        #pragma unroll
        for (int j2 = 0; j2 < 2; ++j2) {
            const int c = (lane << 1) + (j2 << 6);
            const unsigned int lo  = f32_to_bf16_rne(xs[(size_t)c << 8]);
            const unsigned int hi2 = f32_to_bf16_rne(xs[(size_t)(c + 1) << 8]);
            *(unsigned int*)(smem + p * ROWB + (c << 1)) = lo | (hi2 << 16);
        }
    }
    __syncthreads();

    // ---------- Stage 2: K-loop, 36 x 16 WMMAs per wave -----------------------
    const int co_base = wave << 4;
    v8f acc[16] = {};

    union AB { v16bf v; v8bf h[2]; };

    // A row base: K-halves interleaved per 16-bit A 16x32 layout:
    // lanes 0-15 carry K {0..7,16..23}, lanes 16-31 carry K {8..15,24..31}.
    const unsigned short* wrow = Wb + (size_t)(co_base + m) * 1152 + hi * 8;

    #pragma unroll 1
    for (int t = 0; t < 9; ++t) {
        const int kh = t / 3, kw = t - (t / 3) * 3;
        // Warm L0 for the next tap's weight row (global_prefetch_b8).
        __builtin_prefetch(wrow + (t < 8 ? (t + 1) : 8) * 128, 0, 0);
        #pragma unroll
        for (int kc = 0; kc < 4; ++kc) {          // 128 Cin / 32 K per WMMA
            AB a;
            a.h[0] = *(const v8bf*)(wrow + t * 128 + kc * 32);
            a.h[1] = *(const v8bf*)(wrow + t * 128 + kc * 32 + 16);
            const int boff = kc * 64 + hi * 16;   // byte offset of this lane's K run
            #pragma unroll
            for (int h = 0; h < 16; ++h) {        // 16 output rows = 16 N tiles
                const int pos = (h + kh) * 18 + (m + kw);
                AB bm;
                bm.h[0] = *(const v8bf*)(smem + pos * ROWB + boff);
                bm.h[1] = *(const v8bf*)(smem + pos * ROWB + boff + 32);
                acc[h] = __builtin_amdgcn_wmma_f32_16x16x32_bf16(
                    /*neg_a=*/false, a.v, /*neg_b=*/false, bm.v,
                    /*c_mod=*/(short)0, acc[h], /*reuse_a=*/false, /*reuse_b=*/false);
            }
        }
    }

    // ---------- Stage 3: bias + mask epilogue, f32 stores ---------------------
    const v8f bv = *(const v8f*)(bias + co_base + (hi << 3));   // 32B contiguous

    #pragma unroll
    for (int h = 0; h < 16; ++h) {
        const float f = (float)((cMASK[h] >> m) & 1);
        float* orow = out + (((size_t)b * 128 + co_base + (hi << 3)) << 8) + (h << 4) + m;
        #pragma unroll
        for (int j = 0; j < 8; ++j) {
            // C/D layout: VGPR j -> M=j (lanes 0-15) or M=j+8 (lanes 16-31); N = lane&15
            orow[(size_t)j << 8] = (acc[h][j] + bv[j]) * f;
        }
    }
}

// ---- Host entry -------------------------------------------------------------
extern "C" void kernel_launch(void* const* d_in, const int* in_sizes, int n_in,
                              void* d_out, int out_size, void* d_ws, size_t ws_size,
                              hipStream_t stream) {
    const float* x    = (const float*)d_in[0];   // [2048,128,16,16] f32
    const float* W    = (const float*)d_in[1];   // [128,128,3,3]    f32
    const float* bias = (const float*)d_in[2];   // [128]            f32
    float* out        = (float*)d_out;           // [2048,128,16,16] f32
    unsigned short* Wb = (unsigned short*)d_ws;  // 128*1152 bf16 = 294,912 B

    (void)in_sizes; (void)n_in; (void)out_size; (void)ws_size;

    // 88 KB dynamic LDS exceeds the default static cap; raise the limit (idempotent).
    hipFuncSetAttribute((const void*)kagome_conv_wmma_kernel,
                        hipFuncAttributeMaxDynamicSharedMemorySize, LDS_BYTES);

    pack_weights_kernel<<<(128 * 1152 + 255) / 256, 256, 0, stream>>>(W, Wb);
    kagome_conv_wmma_kernel<<<2048, 256, LDS_BYTES, stream>>>(x, Wb, bias, out);
}
